// ScaledDotProductAttention_75462575391105
// MI455X (gfx1250) — matrix operand order repaired and checked by reading, not compiled, not run on hardware
//
#include <hip/hip_runtime.h>
#include <hip/hip_bf16.h>

#define NB      8
#define NH      16
#define S_LEN   1024
#define D_LEN   64
#define BQ      32          // query rows per block
#define SS_STRIDE 1025      // f32 scores row stride (bank-conflict pad)
#define SP_STRIDE 1032      // f16 probs row stride (16B-aligned pad)
#define VT_STRIDE 72        // f16 V-transposed row stride (144B = 9*16B)

// LDS carve offsets (bytes)
#define OFF_SS  0
#define OFF_SP  131200      // 32*1025*4
#define OFF_SQ  197248      // + 32*1032*2
#define OFF_SVT 201344      // + 32*64*2
#define LDS_BYTES 210560    // + 64*72*2

typedef _Float16 h8_t __attribute__((ext_vector_type(8)));
typedef _Float16 v16h __attribute__((ext_vector_type(16)));
typedef float    v8f  __attribute__((ext_vector_type(8)));

// Build 16-bit A/B fragment from two contiguous 8-half (16B) runs.
__device__ __forceinline__ v16h make_frag(const _Float16* p0, const _Float16* p1) {
  h8_t lo = *(const h8_t*)p0;
  h8_t hi = *(const h8_t*)p1;
  v16h a;
#pragma unroll
  for (int i = 0; i < 8; ++i) { a[i] = lo[i]; a[8 + i] = hi[i]; }
  return a;
}

__global__ __launch_bounds__(256) void fa_fwd_attn_kernel(
    const float* __restrict__ gQ, const float* __restrict__ gK,
    const float* __restrict__ gV, const float* __restrict__ gM,
    float* __restrict__ gOut, float* __restrict__ gAttn) {
  extern __shared__ char smem[];
  float*    sS  = (float*)(smem + OFF_SS);    // [32][1025] f32 scores
  _Float16* sP  = (_Float16*)(smem + OFF_SP); // [32][1032] f16 probs
  _Float16* sQ  = (_Float16*)(smem + OFF_SQ); // [32][64]   f16 Q tile
  _Float16* sVt = (_Float16*)(smem + OFF_SVT);// [64][72]   f16 V^T chunk

  const int tid  = threadIdx.x;
  const int lane = tid & 31;
  const int w    = tid >> 5;
  const int lo16 = lane & 15;
  const bool hi  = lane >= 16;

  const int bid = blockIdx.x;
  const int qt  = bid & 31;          // S/BQ = 32 q-tiles per (b,h)
  const int bh  = bid >> 5;
  const int q0  = qt * BQ;

  const size_t qkvBase = (size_t)bh * S_LEN * D_LEN;
  const float* Qp = gQ + qkvBase + (size_t)q0 * D_LEN;
  const float* Kp = gK + qkvBase;
  const float* Vp = gV + qkvBase;
  float* attnP = gAttn + (size_t)bh * S_LEN * S_LEN;

  // ---------------- Phase 1: stage Q tile (f32 -> f16) ----------------
  for (int i = tid; i < BQ * D_LEN; i += 256)
    sQ[i] = (_Float16)Qp[i];
  __syncthreads();

  // ---------------- Phase 2: scores = Q K^T / T + mask ----------------
  {
    const int m0   = (w & 1) * 16;          // row tile
    const int arow = m0 + lo16;
    const int klo  = hi ? 8 : 0;            // ISA 16-bit A layout
    v16h a0 = make_frag(&sQ[arow * D_LEN +  0 + klo], &sQ[arow * D_LEN + 16 + klo]);
    v16h a1 = make_frag(&sQ[arow * D_LEN + 32 + klo], &sQ[arow * D_LEN + 48 + klo]);

    for (int kt = (w >> 1); kt < S_LEN / 16; kt += 4) {
      const int n0     = kt * 16;
      const int keyrow = n0 + lo16;
      const float* kp  = Kp + (size_t)keyrow * D_LEN + (hi ? 8 : 0);
      v16h b0, b1;
#pragma unroll
      for (int i = 0; i < 16; ++i) b0[i] = (_Float16)kp[i + (i & 8)];
#pragma unroll
      for (int i = 0; i < 16; ++i) b1[i] = (_Float16)kp[32 + i + (i & 8)];

      v8f acc = {};
      acc = __builtin_amdgcn_wmma_f32_16x16x32_f16(false, a0, false, b0,
                                                   (short)0, acc, false, false);
      acc = __builtin_amdgcn_wmma_f32_16x16x32_f16(false, a1, false, b1,
                                                   (short)0, acc, false, false);

      const int mbase = m0 + (hi ? 8 : 0);  // ISA C/D layout
      const int col   = n0 + lo16;
#pragma unroll
      for (int r = 0; r < 8; ++r) {
        const int row = mbase + r;
        float sval = acc[r] * 0.125f + gM[(size_t)(q0 + row) * S_LEN + col];
        sS[row * SS_STRIDE + col] = sval;
      }
    }
  }
  __syncthreads();

  // ---------------- Phase 3: softmax + write attn + f16 probs ----------------
  for (int rr = w * 4; rr < w * 4 + 4; ++rr) {
    float mx = -3.0e38f;
    for (int c = lane; c < S_LEN; c += 32)
      mx = fmaxf(mx, sS[rr * SS_STRIDE + c]);
#pragma unroll
    for (int off = 16; off > 0; off >>= 1)
      mx = fmaxf(mx, __shfl_xor(mx, off, 32));

    float sum = 0.f;
    for (int c = lane; c < S_LEN; c += 32) {
      float e = __expf(sS[rr * SS_STRIDE + c] - mx);
      sS[rr * SS_STRIDE + c] = e;
      sum += e;
    }
#pragma unroll
    for (int off = 16; off > 0; off >>= 1)
      sum += __shfl_xor(sum, off, 32);
    const float inv = 1.0f / sum;

    float* arow = attnP + (size_t)(q0 + rr) * S_LEN;
    for (int c = lane; c < S_LEN; c += 32) {
      float p = sS[rr * SS_STRIDE + c] * inv;
      arow[c] = p;                            // coalesced 128B/wave
      sP[rr * SP_STRIDE + c] = (_Float16)p;
    }
  }
  __syncthreads();

  // ---------------- Phase 4: out = P @ V ----------------
  const int m0   = (w & 1) * 16;
  const int n0   = (w >> 1) * 16;             // 4 col tiles x 2 row tiles = 8 waves
  const int klo  = hi ? 8 : 0;
  const int arow = m0 + lo16;
  const int dcol = n0 + lo16;
  const int kk   = hi ? 8 : 0;
  v8f oacc = {};

  for (int kb = 0; kb < S_LEN; kb += 64) {
    // stage 64-key V chunk transposed: sVt[d][key]
    for (int i = tid; i < 64 * 64; i += 256) {
      int key = i >> 6;
      int dd  = i & 63;
      sVt[dd * VT_STRIDE + key] = (_Float16)Vp[(size_t)(kb + key) * D_LEN + dd];
    }
    __syncthreads();
#pragma unroll
    for (int c = 0; c < 2; ++c) {
      const int ka = kb + c * 32;             // key offset in probs
      const int kl = c * 32;                  // key offset in sVt chunk
      v16h a = make_frag(&sP[arow * SP_STRIDE + ka + klo],
                         &sP[arow * SP_STRIDE + ka + 16 + klo]);
      const _Float16* vp = &sVt[dcol * VT_STRIDE + kl + kk];
      v16h b = make_frag(vp, vp + 16);
      oacc = __builtin_amdgcn_wmma_f32_16x16x32_f16(false, a, false, b,
                                                    (short)0, oacc, false, false);
    }
    __syncthreads();
  }

  // ---------------- store out tile ----------------
  float* op = gOut + qkvBase + (size_t)q0 * D_LEN;
  const int mbase = m0 + (hi ? 8 : 0);
#pragma unroll
  for (int r = 0; r < 8; ++r)
    op[(size_t)(mbase + r) * D_LEN + n0 + lo16] = oacc[r];
}

extern "C" void kernel_launch(void* const* d_in, const int* in_sizes, int n_in,
                              void* d_out, int out_size, void* d_ws, size_t ws_size,
                              hipStream_t stream) {
  const float* q = (const float*)d_in[0];
  const float* k = (const float*)d_in[1];
  const float* v = (const float*)d_in[2];
  const float* m = (const float*)d_in[3];
  float* out  = (float*)d_out;
  float* attn = out + (size_t)NB * NH * S_LEN * D_LEN;  // tuple order: (out, attn)

  dim3 grid(NB * NH * (S_LEN / BQ));   // 4096 blocks
  fa_fwd_attn_kernel<<<grid, 256, LDS_BYTES, stream>>>(q, k, v, m, out, attn);
}
